// NTT_zkp_57853209477456
// MI455X (gfx1250) — 8-bit matrix operand order repaired and checked by reading, compiled, not run on hardware
//
#include <hip/hip_runtime.h>
#include <stdint.h>

#define PRI 2013265921u
static constexpr unsigned long long MAGIC = 0xFFFFFFFFFFFFFFFFull / PRI;
static constexpr int  LOGN   = 22;
static constexpr unsigned NN = 1u << LOGN;
static constexpr unsigned NMASK = NN - 1u;
static constexpr unsigned HALF_N = NN >> 1;
static constexpr int  LOGT   = 11;
static constexpr int  TN     = 1 << LOGT;
static constexpr int  TILE   = 16;
static constexpr int  PAD    = 8;
static constexpr int  STR    = TN + PAD;
static constexpr int  THREADS= 256;
static constexpr int  BATCH  = 8;
static constexpr int  NTW    = 2032;

constexpr unsigned modpow_c(unsigned long long b, unsigned long long e) {
  unsigned long long r = 1, x = b % PRI;
  while (e) { if (e & 1) r = r * x % PRI; x = x * x % PRI; e >>= 1; }
  return (unsigned)r;
}
static constexpr unsigned NINV = modpow_c(modpow_c(2, LOGN), PRI - 2);
static constexpr unsigned C8W[7] = {
  modpow_c(2, 0),  modpow_c(2, 8),  modpow_c(2, 16), modpow_c(2, 24),
  modpow_c(2, 32), modpow_c(2, 40), modpow_c(2, 48)
};
__device__ constexpr int IW[7][4] = {
  {0,-1,-1,-1},{0,1,-1,-1},{0,1,2,-1},{0,1,2,3},{1,2,3,-1},{2,3,-1,-1},{3,-1,-1,-1}};
__device__ constexpr int JW[7][4] = {
  {0,-1,-1,-1},{1,0,-1,-1},{2,1,0,-1},{3,2,1,0},{3,2,1,-1},{3,2,-1,-1},{3,-1,-1,-1}};
__device__ constexpr int BR4[16] = {0,8,4,12,2,10,6,14,1,9,5,13,3,11,7,15};

typedef __attribute__((ext_vector_type(8))) int v8i;

__device__ __forceinline__ unsigned reduce64(unsigned long long t) {
  unsigned long long q = __umul64hi(t, MAGIC);
  unsigned long long r = t - q * (unsigned long long)PRI;
  if (r >= 2ull * PRI) r -= 2ull * PRI;
  if (r >= PRI)        r -= PRI;
  return (unsigned)r;
}
__device__ __forceinline__ unsigned mulmod(unsigned a, unsigned b) {
  return reduce64((unsigned long long)a * b);
}
__device__ __forceinline__ unsigned addm(unsigned a, unsigned b) {
  unsigned s = a + b; return (s >= PRI) ? s - PRI : s;
}
__device__ __forceinline__ unsigned subm(unsigned a, unsigned b) {
  return (a >= b) ? a - b : a + PRI - b;
}
__device__ __forceinline__ unsigned brev11(unsigned x) { return __brev(x) >> (32 - LOGT); }

__device__ __forceinline__ void async_g2l_b32(unsigned lds_byte_off, unsigned long long gaddr) {
  asm volatile("global_load_async_to_lds_b32 %0, %1, off"
               :: "v"(lds_byte_off), "v"(gaddr) : "memory");
}
__device__ __forceinline__ void wait_async0() {
  asm volatile("s_wait_asynccnt 0" ::: "memory");
}

__device__ __forceinline__ void preload_stage_tw(unsigned* sTw, const unsigned* tab, int tid) {
  for (int i = tid; i < NTW; i += THREADS) {
    int v  = i + 16;
    int lm = 32 - __clz(v);
    int t  = v - (1 << (lm - 1));
    sTw[i] = tab[(unsigned)t << (LOGN - lm)];
  }
}

__device__ void ntt2048_tile(unsigned* data, unsigned* sW16, const unsigned* sTw,
                             const unsigned* tab, const unsigned* ctab, int j2base) {
  const int tid = threadIdx.x;
  if (tid < 16) {
    unsigned t = (unsigned)tid;
    sW16[t] = (t < 8) ? tab[(NN >> 4) * t] : PRI - tab[(NN >> 4) * (t - 8)];
  }
  __syncthreads();

  const int lane = tid & 31;
  const int wave = tid >> 5;
  const int col  = lane & 15;
  const int h    = lane >> 4;
  unsigned* colp = data + col * STR;

  v8i Af[7];
  #pragma unroll
  for (int w = 0; w < 7; ++w) {
    #pragma unroll
    for (int v = 0; v < 8; ++v) {
      unsigned word = 0;
      #pragma unroll
      for (int b = 0; b < 4; ++b) {
        int K = ((v < 4 ? h : 2 + h) << 4) + ((v & 3) << 2) + b;
        int g = K >> 4, c = K & 15;
        int i = IW[w][g];
        unsigned byte = 0;
        if (i >= 0) {
          unsigned m = sW16[(col * BR4[c]) & 15];
          byte = (m >> (8 * i)) & 0xFFu;
        }
        word |= byte << (8 * b);
      }
      Af[w][v] = (int)word;
    }
  }

  for (int blk = wave; blk < TN / 16; blk += 8) {
    unsigned X[16];
    #pragma unroll
    for (int k = 0; k < 16; ++k) X[k] = colp[blk * 16 + k];
    if (ctab) {
      #pragma unroll
      for (int k = 0; k < 16; ++k) {
        unsigned j1 = brev11((unsigned)(blk * 16 + k));
        X[k] = mulmod(X[k], ctab[(unsigned)(j2base + col) + (j1 << LOGT)]);
      }
    }
    unsigned long long acc[8] = {0,0,0,0,0,0,0,0};
    #pragma unroll
    for (int w = 0; w < 7; ++w) {
      v8i Bf;
      #pragma unroll
      for (int v = 0; v < 8; ++v) {
        int g = (v < 4) ? h : (2 + h);
        int j = JW[w][g];
        int kbase = 4 * (v & 3);
        unsigned word = 0;
        if (j >= 0) {
          #pragma unroll
          for (int b = 0; b < 4; ++b)
            word |= ((X[kbase + b] >> (8 * j)) & 0xFFu) << (8 * b);
        }
        Bf[v] = (int)word;
      }
      v8i Cz = {};
      v8i D = __builtin_amdgcn_wmma_i32_16x16x64_iu8(false, Af[w], false, Bf, Cz,
                                                     false, false);
      #pragma unroll
      for (int p = 0; p < 8; ++p)
        acc[p] += (unsigned long long)(unsigned)D[p] * (unsigned long long)C8W[w];
    }
    #pragma unroll
    for (int p = 0; p < 8; ++p)
      colp[blk * 16 + p + 8 * h] = reduce64(acc[p]);
  }
  __syncthreads();

  const int colS = tid & 15;
  const int sub  = tid >> 4;
  unsigned* cp = data + colS * STR;
  for (int lm = 5; lm <= LOGT; ++lm) {
    const int half = 1 << (lm - 1);
    const unsigned* stw = sTw + (half - 16);
    for (int it = 0; it < (TN / 2) / 16; ++it) {
      int bi = sub + (it << 4);
      int t  = bi & (half - 1);
      int jj = ((bi >> (lm - 1)) << lm) + t;
      unsigned u = cp[jj];
      unsigned v = mulmod(cp[jj + half], stw[t]);
      cp[jj]        = addm(u, v);
      cp[jj + half] = subm(u, v);
    }
    __syncthreads();
  }
}

static constexpr size_t SMEM_WORDS = 16 + NTW + (size_t)TILE * STR;

__global__ __launch_bounds__(THREADS)
void ntt_pass1(const int* __restrict__ in, const int* tw, const int* itw,
               const int* coset, const int* is_intt_p, const int* is_coset_p,
               unsigned* __restrict__ T) {
  extern __shared__ unsigned smem[];
  unsigned* sW16 = smem;
  unsigned* sTw  = smem + 16;
  unsigned* data = smem + 16 + NTW;
  const int wg = blockIdx.x;
  const int b  = wg >> 7;
  const int j2base = (wg & 127) << 4;
  const int tid = threadIdx.x;
  const int is_intt  = is_intt_p[0];
  const int is_coset = is_coset_p[0];
  const unsigned* tab  = (const unsigned*)(is_intt ? itw : tw);
  const unsigned* ctab = (!is_intt && is_coset) ? (const unsigned*)coset : nullptr;

  const unsigned ldsBase = (unsigned)(uintptr_t)(void*)data;
  const int* src = in + ((size_t)b << LOGN);
  for (int it = 0; it < (TN * TILE) / THREADS; ++it) {
    int idx = it * THREADS + tid;
    int c = idx & 15, j1 = idx >> 4;
    unsigned off = ldsBase + ((unsigned)c * STR + brev11((unsigned)j1)) * 4u;
    async_g2l_b32(off, (unsigned long long)(uintptr_t)(src + (size_t)j1 * TN + j2base + c));
  }
  preload_stage_tw(sTw, tab, tid);
  wait_async0();
  __syncthreads();

  ntt2048_tile(data, sW16, sTw, tab, ctab, j2base);

  for (int it = 0; it < (TN * TILE) / THREADS; ++it) {
    int idx = it * THREADS + tid;
    int c = idx & 15, k1 = idx >> 4;
    unsigned val = data[c * STR + k1];
    unsigned e = ((unsigned)(j2base + c) * (unsigned)k1) & NMASK;
    unsigned f = (e < HALF_N) ? tab[e] : PRI - tab[e - HALF_N];
    T[((size_t)b << LOGN) + ((size_t)k1 << LOGT) + (unsigned)(j2base + c)] = mulmod(val, f);
  }
}

__global__ __launch_bounds__(THREADS)
void ntt_pass2(const unsigned* __restrict__ T, const int* tw, const int* itw,
               const int* icoset, const int* is_intt_p, const int* is_coset_p,
               unsigned* __restrict__ out) {
  extern __shared__ unsigned smem[];
  unsigned* sW16 = smem;
  unsigned* sTw  = smem + 16;
  unsigned* data = smem + 16 + NTW;
  const int wg = blockIdx.x;
  const int b  = wg >> 7;
  const int k1base = (wg & 127) << 4;
  const int tid = threadIdx.x;
  const int is_intt  = is_intt_p[0];
  const int is_coset = is_coset_p[0];
  const unsigned* tab = (const unsigned*)(is_intt ? itw : tw);

  const unsigned* Tb = T + ((size_t)b << LOGN) + ((size_t)k1base << LOGT);
  const unsigned ldsBase = (unsigned)(uintptr_t)(void*)data;
  for (int it = 0; it < (TN * TILE) / THREADS; ++it) {
    int idx = it * THREADS + tid;
    int row = idx >> LOGT, j2 = idx & (TN - 1);
    unsigned off = ldsBase + ((unsigned)row * STR + brev11((unsigned)j2)) * 4u;
    async_g2l_b32(off, (unsigned long long)(uintptr_t)(Tb + idx));
  }
  preload_stage_tw(sTw, tab, tid);
  wait_async0();
  __syncthreads();

  ntt2048_tile(data, sW16, sTw, tab, nullptr, 0);

  for (int it = 0; it < (TN * TILE) / THREADS; ++it) {
    int idx = it * THREADS + tid;
    int row = idx & 15, k2 = idx >> 4;
    unsigned val = data[row * STR + k2];
    unsigned kg  = (unsigned)(k1base + row) + ((unsigned)k2 << LOGT);
    if (is_intt) {
      val = mulmod(val, NINV);
      if (is_coset) val = mulmod(val, (unsigned)icoset[kg]);
    }
    out[((size_t)b << LOGN) + kg] = val;
  }
}

extern "C" void kernel_launch(void* const* d_in, const int* in_sizes, int n_in,
                              void* d_out, int out_size, void* d_ws, size_t ws_size,
                              hipStream_t stream) {
  const int* in        = (const int*)d_in[0];
  const int* tw        = (const int*)d_in[1];
  const int* itw       = (const int*)d_in[2];
  const int* coset     = (const int*)d_in[3];
  const int* icoset    = (const int*)d_in[4];
  const int* is_intt_p  = (const int*)d_in[6];
  const int* is_coset_p = (const int*)d_in[7];
  unsigned* T   = (unsigned*)d_ws;
  unsigned* out = (unsigned*)d_out;

  const size_t smem = SMEM_WORDS * sizeof(unsigned);
  const dim3 grid(BATCH * (TN / TILE));
  const dim3 blk(THREADS);

  ntt_pass1<<<grid, blk, smem, stream>>>(in, tw, itw, coset, is_intt_p, is_coset_p, T);
  ntt_pass2<<<grid, blk, smem, stream>>>(T, tw, itw, icoset, is_intt_p, is_coset_p, out);
}
